// LSTM_62354335203966
// MI455X (gfx1250) — matrix operand order repaired and checked by reading, not compiled, not run on hardware
//
#include <hip/hip_runtime.h>
#include <hip/hip_bf16.h>
#include <stdint.h>

// ---------------- problem constants ----------------
#define T_STEPS 512
#define BATCH   128
#define INP     256
#define HID     512
#define GATES   2048          // 4*HID
#define NOUT    5
#define KT1     24            // (INP+HID)/32  : K-tiles for layer-1 GEMM
#define KT2     32            // (2*HID)/32    : K-tiles for layer-2 GEMM
#define NWG     128           // 8 groups x 16 WGs
#define GRP_WGS 16            // WGs per batch-row group (independent pipelines)
#define TPB     256           // 8 waves of 32

typedef __attribute__((ext_vector_type(16))) __bf16 v16bf;
typedef __attribute__((ext_vector_type(8)))  float  v8f;
typedef int v4i_vs __attribute__((vector_size(16)));   // matches builtin param type

#define AS1 __attribute__((address_space(1)))
#define AS3 __attribute__((address_space(3)))

#if defined(__AMDGCN__) && __has_builtin(__builtin_amdgcn_global_load_async_to_lds_b128)
#define HAVE_ASYNC_LDS 1
#else
#define HAVE_ASYNC_LDS 0
#endif

// ---------------- helpers ----------------
__device__ __forceinline__ unsigned short f32_to_bf16(float f) {
  unsigned u = __float_as_uint(f);
  u += 0x7FFFu + ((u >> 16) & 1u);      // round-to-nearest-even
  return (unsigned short)(u >> 16);
}
__device__ __forceinline__ float sigmoidf_(float x) {
  return 1.0f / (1.0f + __expf(-x));
}

// Store one scalar into the WMMA bf16 A-fragment layout (16x32 tiles).
// ISA 7.12.2: lanes 0-15 hold K in {0..7,16..23}, lanes 16-31 hold {8..15,24..31}.
__device__ __forceinline__ void store_packA(unsigned short* p, int b, int j, float v) {
  int mt = b >> 4, m = b & 15;
  int kt = j >> 5, kk = j & 31;
  int grp  = kk >> 3;
  int lane = ((grp & 1) << 4) | m;
  int slot = ((grp >> 1) << 3) | (kk & 7);
  p[(((size_t)(mt * 16 + kt) * 32) + lane) * 16 + slot] = f32_to_bf16(v);
}

// async (ASYNCcnt-tracked) copy of one 8 KB x-slice (512 uint4) into LDS
__device__ __forceinline__ void async_copy_x(const uint4* g, uint4* l, int tid) {
#if HAVE_ASYNC_LDS
  #pragma unroll
  for (int i = tid; i < 512; i += TPB) {
    __builtin_amdgcn_global_load_async_to_lds_b128(
        (AS1 v4i_vs*)(g + i), (AS3 v4i_vs*)(l + i), 0, 0);
  }
#else
  (void)g; (void)l; (void)tid;
#endif
}
__device__ __forceinline__ void wait_async_le2() {
#if HAVE_ASYNC_LDS
#if __has_builtin(__builtin_amdgcn_s_wait_asynccnt)
  __builtin_amdgcn_s_wait_asynccnt(2);
#else
  asm volatile("s_wait_asynccnt 0x2" ::: "memory");
#endif
#endif
}

// ---------------- pre-pass: pack X into bf16 A-fragments ----------------
// Xp layout: [t][mt(8)][kt(8)][lane(32)][slot(16)]
__global__ void pack_x_kernel(const float* __restrict__ x, unsigned short* __restrict__ xp) {
  const long long total = (long long)T_STEPS * BATCH * INP;
  for (long long d = (long long)blockIdx.x * blockDim.x + threadIdx.x; d < total;
       d += (long long)gridDim.x * blockDim.x) {
    int slot = (int)(d & 15);
    int lane = (int)((d >> 4) & 31);
    int kt   = (int)((d >> 9) & 7);
    int mt   = (int)((d >> 12) & 7);
    int t    = (int)(d >> 15);
    int grp  = ((slot >> 3) << 1) | (lane >> 4);
    int kk   = (grp << 3) | (slot & 7);
    int b    = (mt << 4) | (lane & 15);
    int k    = (kt << 5) | kk;
    xp[d] = f32_to_bf16(x[((size_t)t * BATCH + b) * INP + k]);
  }
}

// ---------------- pre-pass: pack W^T (concat [Wi;Wh] along K) into B-fragments ----
// wp layout: [nt(128)][kt(ktn)][lane(32)][slot(16)]
__global__ void pack_w_kernel(const float* __restrict__ Wi, const float* __restrict__ Wh,
                              unsigned short* __restrict__ wp,
                              int ktn, int ksplit, int wi_cols, int wh_cols) {
  const long long total = (long long)128 * ktn * 512;
  for (long long d = (long long)blockIdx.x * blockDim.x + threadIdx.x; d < total;
       d += (long long)gridDim.x * blockDim.x) {
    int slot = (int)(d & 15);
    int lane = (int)((d >> 4) & 31);
    long long r = d >> 9;
    int kt = (int)(r % ktn);
    int nt = (int)(r / ktn);
    int kk = ((slot >> 3) << 4) | ((lane >> 4) << 3) | (slot & 7);
    int n  = (nt << 4) | (lane & 15);
    int k  = (kt << 5) + kk;
    float v = (k < ksplit) ? Wi[(size_t)n * wi_cols + k]
                           : Wh[(size_t)n * wh_cols + (k - ksplit)];
    wp[d] = f32_to_bf16(v);
  }
}

__global__ void bias_init_kernel(const float* bi1, const float* bh1,
                                 const float* bi2, const float* bh2,
                                 float* bias1, float* bias2) {
  int n = blockIdx.x * blockDim.x + threadIdx.x;
  if (n < GATES) { bias1[n] = bi1[n] + bh1[n]; bias2[n] = bi2[n] + bh2[n]; }
}

__global__ void zero_states_kernel(uint4* states, long long n4, unsigned* syncv) {
  uint4 z = {0u, 0u, 0u, 0u};
  long long stride = (long long)gridDim.x * blockDim.x;
  for (long long i = (long long)blockIdx.x * blockDim.x + threadIdx.x; i < n4; i += stride)
    states[i] = z;
  for (long long i = (long long)blockIdx.x * blockDim.x + threadIdx.x; i < 512; i += stride)
    syncv[i] = 0u;
}

// ---------------- persistent recurrent kernel ----------------
// 8 independent groups (one per 16 batch rows); 16 WGs per group; 1 output tile
// per wave per GEMM; weight B fragments re-read from W1p/W2p every time step.
__global__ __launch_bounds__(TPB, 1) void lstm_persistent_kernel(
    const uint4* __restrict__ Xp,
    const uint4* __restrict__ W1p,
    const uint4* __restrict__ W2p,
    const float* __restrict__ bias1,
    const float* __restrict__ bias2,
    unsigned short* __restrict__ h1p,
    unsigned short* __restrict__ h2p,
    float* __restrict__ gates1,
    float* __restrict__ gates2,
    float* __restrict__ sel,
    const int* __restrict__ len,
    unsigned* syncv)
{
  __shared__ __align__(32) uint4 xbuf[2][512];   // double-buffered x slice (2 x 8 KB)
  __shared__ __align__(32) uint4 hbuf[2048];     // h operand stage (32 KB)

  const int tid  = threadIdx.x;
  const int wg   = blockIdx.x;
  const int mt   = wg >> 4;                      // group / M-tile row (0..7)
  const int ntb  = wg & 15;                      // N-tile block within group
  const int wave = tid >> 5;
  const int lane = tid & 31;
  const int nt   = ntb * 8 + wave;               // this wave's N-tile (0..127)
  const int ncol = nt * 16 + (lane & 15);        // D-layout column
  const int grow = mt * 16 + ((lane >> 4) << 3); // first D-layout row

  const uint4* h1q = (const uint4*)h1p + (size_t)mt * 1024;
  const uint4* h2q = (const uint4*)h2p + (size_t)mt * 1024;

  // group-local elementwise identity: 4096 threads cover 16 rows x 512 cols x2
  const int gl = ntb * TPB + tid;                // 0..4095
  const int e  = gl * 2;
  const int rb = e >> 9;                         // row in group 0..15
  const int ej = e & 511;
  const int b  = mt * 16 + rb;
  const int lv = len[b];
  float creg1[2] = {0.f, 0.f};                   // cell state lives in registers
  float creg2[2] = {0.f, 0.f};

  const float bv1 = bias1[ncol];
  const float bv2 = bias2[ncol];

  volatile unsigned* cnt = syncv + (size_t)mt * 64;
  volatile unsigned* gen = cnt + 16;
  auto gsync = [&]() {                           // 16-WG group barrier
    __syncthreads();
    if (tid == 0) {
      __threadfence();
      unsigned g = *gen;
      if (atomicAdd((unsigned*)cnt, 1u) == (unsigned)(GRP_WGS - 1)) {
        *cnt = 0u;
        __threadfence();
        atomicAdd((unsigned*)gen, 1u);
      } else {
        while (*gen == g) { __builtin_amdgcn_s_sleep(1); }
      }
      __threadfence();
    }
    __syncthreads();
  };

#if HAVE_ASYNC_LDS
  // pre-fill the x pipeline: x(0) and x(1)
  async_copy_x(Xp + (size_t)(0 * 8 + mt) * 512, &xbuf[0][0], tid);
  async_copy_x(Xp + (size_t)(1 * 8 + mt) * 512, &xbuf[1][0], tid);
#endif

  for (int t = 0; t < T_STEPS; ++t) {
    const int par = t & 1;

    // ---- stage A1: h1(t-1) slice -> LDS; x(t) already async-prefetched ----
    for (int i = tid; i < 1024; i += TPB) hbuf[i] = h1q[i];
#if !HAVE_ASYNC_LDS
    {
      const uint4* xs = Xp + ((size_t)t * 8 + mt) * 512;
      for (int i = tid; i < 512; i += TPB) xbuf[par][i] = xs[i];
    }
#endif
    wait_async_le2();                  // oldest prefetch (x(t)) complete
    __syncthreads();

    // ---- GEMM1: gates1 = bias1 + [x(t)|h1] @ [W_ih1;W_hh1]^T ----
    {
      v8f acc;
      #pragma unroll
      for (int r = 0; r < 8; ++r) acc[r] = bv1;
      const v16bf* xv = (const v16bf*)xbuf[par];
      const v16bf* hv = (const v16bf*)hbuf;
      const v16bf* Bf = ((const v16bf*)W1p) + ((size_t)nt * KT1 * 32 + lane);
      #pragma unroll
      for (int kt = 0; kt < 8; ++kt)
        acc = __builtin_amdgcn_wmma_f32_16x16x32_bf16(false, xv[kt * 32 + lane], false,
                                                      Bf[(size_t)kt * 32], (short)0, acc,
                                                      false, false);
      #pragma unroll
      for (int kt = 8; kt < KT1; ++kt)
        acc = __builtin_amdgcn_wmma_f32_16x16x32_bf16(false, hv[(kt - 8) * 32 + lane], false,
                                                      Bf[(size_t)kt * 32], (short)0, acc,
                                                      false, false);
      float* gp = gates1 + (size_t)grow * GATES + ncol;
      #pragma unroll
      for (int r = 0; r < 8; ++r) gp[(size_t)r * GATES] = acc[r];
    }
    gsync();                                        // SYNC A

#if HAVE_ASYNC_LDS
    // xbuf[par] is free now (all waves past barrier): prefetch x(t+2) into it
    if (t + 2 < T_STEPS)
      async_copy_x(Xp + ((size_t)(t + 2) * 8 + mt) * 512, &xbuf[par][0], tid);
#endif

    // ---- cell 1 (i,f,g,o) + extra tanh; h1 written in A-fragment layout ----
    {
      const float* gb = gates1 + (size_t)b * GATES;
      #pragma unroll
      for (int q = 0; q < 2; ++q) {
        int j = ej + q;
        float gi = gb[j], gf = gb[512 + j], gg = gb[1024 + j], go = gb[1536 + j];
        float cn = sigmoidf_(gf) * creg1[q] + sigmoidf_(gi) * tanhf(gg);
        float hn = tanhf(sigmoidf_(go) * tanhf(cn));   // extra tanh per reference
        creg1[q] = cn;
        store_packA(h1p, b, j, hn);
      }
    }
    gsync();                                        // SYNC B

    // ---- stage A2 = [h1(t) | h2(t-1)] ----
    for (int i = tid; i < 1024; i += TPB) { hbuf[i] = h1q[i]; hbuf[i + 1024] = h2q[i]; }
    __syncthreads();

    // ---- GEMM2: gates2 = bias2 + [h1|h2] @ [W_ih2;W_hh2]^T ----
    {
      v8f acc;
      #pragma unroll
      for (int r = 0; r < 8; ++r) acc[r] = bv2;
      const v16bf* hv = (const v16bf*)hbuf;
      const v16bf* Bf = ((const v16bf*)W2p) + ((size_t)nt * KT2 * 32 + lane);
      #pragma unroll
      for (int kt = 0; kt < KT2; ++kt)
        acc = __builtin_amdgcn_wmma_f32_16x16x32_bf16(false, hv[kt * 32 + lane], false,
                                                      Bf[(size_t)kt * 32], (short)0, acc,
                                                      false, false);
      float* gp = gates2 + (size_t)grow * GATES + ncol;
      #pragma unroll
      for (int r = 0; r < 8; ++r) gp[(size_t)r * GATES] = acc[r];
    }
    gsync();                                        // SYNC C

    // ---- cell 2 + select outputs[length[b], b]  (no trailing barrier needed) ----
    {
      const float* gb = gates2 + (size_t)b * GATES;
      #pragma unroll
      for (int q = 0; q < 2; ++q) {
        int j = ej + q;
        float gi = gb[j], gf = gb[512 + j], gg = gb[1024 + j], go = gb[1536 + j];
        float cn = sigmoidf_(gf) * creg2[q] + sigmoidf_(gi) * tanhf(gg);
        float hn = tanhf(sigmoidf_(go) * tanhf(cn));
        creg2[q] = cn;
        store_packA(h2p, b, j, hn);
        if (lv == t) sel[(size_t)b * HID + j] = hn;
      }
    }
  }
}

// ---------------- tail: logits + log_softmax ----------------
__global__ void head_kernel(const float* __restrict__ sel, const float* __restrict__ Wout,
                            const float* __restrict__ bout, float* __restrict__ out) {
  int b = threadIdx.x;
  if (b >= BATCH) return;
  float acc[NOUT];
  #pragma unroll
  for (int o = 0; o < NOUT; ++o) acc[o] = bout[o];
  const float* s = sel + (size_t)b * HID;
  for (int j = 0; j < HID; ++j) {
    float sv = s[j];
    #pragma unroll
    for (int o = 0; o < NOUT; ++o) acc[o] += sv * Wout[(size_t)o * HID + j];
  }
  float m = acc[0];
  #pragma unroll
  for (int o = 1; o < NOUT; ++o) m = fmaxf(m, acc[o]);
  float sum = 0.f;
  #pragma unroll
  for (int o = 0; o < NOUT; ++o) sum += __expf(acc[o] - m);
  float lse = __logf(sum);
  #pragma unroll
  for (int o = 0; o < NOUT; ++o) out[(size_t)b * NOUT + o] = acc[o] - m - lse;
}

// ---------------- host-side launcher ----------------
extern "C" void kernel_launch(void* const* d_in, const int* in_sizes, int n_in,
                              void* d_out, int out_size, void* d_ws, size_t ws_size,
                              hipStream_t stream) {
  (void)in_sizes; (void)n_in; (void)out_size; (void)ws_size;
  const float* x     = (const float*)d_in[0];
  const int*   len   = (const int*)  d_in[1];
  const float* W_ih1 = (const float*)d_in[2];
  const float* W_hh1 = (const float*)d_in[3];
  const float* b_ih1 = (const float*)d_in[4];
  const float* b_hh1 = (const float*)d_in[5];
  const float* W_ih2 = (const float*)d_in[6];
  const float* W_hh2 = (const float*)d_in[7];
  const float* b_ih2 = (const float*)d_in[8];
  const float* b_hh2 = (const float*)d_in[9];
  const float* W_out = (const float*)d_in[10];
  const float* b_out = (const float*)d_in[11];

  char* ws = (char*)d_ws;
  size_t off = 0;
  auto carve = [&](size_t bytes) -> char* {
    char* p = ws + off;
    off += (bytes + 255) & ~(size_t)255;
    return p;
  };
  unsigned short* Xp    = (unsigned short*)carve((size_t)T_STEPS * BATCH * INP * 2); // 32 MB
  unsigned short* W1p   = (unsigned short*)carve((size_t)GATES * (INP + HID) * 2);   // 3 MB
  unsigned short* W2p   = (unsigned short*)carve((size_t)GATES * (2 * HID) * 2);     // 4 MB
  float*          bias1 = (float*)carve(GATES * 4);
  float*          bias2 = (float*)carve(GATES * 4);
  char* states_base = ws + off;                       // h1p|h2p contiguous for zeroing
  unsigned short* h1p = (unsigned short*)carve((size_t)BATCH * HID * 2);
  unsigned short* h2p = (unsigned short*)carve((size_t)BATCH * HID * 2);
  long long states_u4 = ((long long)BATCH * HID * 4) / 16;                           // 16384
  float*    gates1 = (float*)carve((size_t)BATCH * GATES * 4);
  float*    gates2 = (float*)carve((size_t)BATCH * GATES * 4);
  float*    sel    = (float*)carve((size_t)BATCH * HID * 4);
  unsigned* syncv  = (unsigned*)carve(512 * 4);

  hipLaunchKernelGGL(pack_x_kernel, dim3(2048), dim3(TPB), 0, stream, x, Xp);
  hipLaunchKernelGGL(pack_w_kernel, dim3(1024), dim3(TPB), 0, stream,
                     W_ih1, W_hh1, W1p, KT1, INP, INP, HID);
  hipLaunchKernelGGL(pack_w_kernel, dim3(1024), dim3(TPB), 0, stream,
                     W_ih2, W_hh2, W2p, KT2, HID, HID, HID);
  hipLaunchKernelGGL(bias_init_kernel, dim3(8), dim3(TPB), 0, stream,
                     b_ih1, b_hh1, b_ih2, b_hh2, bias1, bias2);
  hipLaunchKernelGGL(zero_states_kernel, dim3(64), dim3(TPB), 0, stream,
                     (uint4*)states_base, states_u4, syncv);

  hipLaunchKernelGGL(lstm_persistent_kernel, dim3(NWG), dim3(TPB), 0, stream,
                     (const uint4*)Xp, (const uint4*)W1p, (const uint4*)W2p,
                     bias1, bias2, h1p, h2p, gates1, gates2, sel, len, syncv);

  hipLaunchKernelGGL(head_kernel, dim3(1), dim3(128), 0, stream,
                     sel, W_out, b_out, (float*)d_out);
}
